// ManualAttention_83863531421996
// MI455X (gfx1250) — matrix operand order repaired and checked by reading, not compiled, not run on hardware
//
#include <hip/hip_runtime.h>

#define D_MODEL 768
#define N_HEADS 12
#define D_HEAD  64
#define SEQ     4096
#define BATCH   2
#define ROWS    (BATCH * SEQ)     // 8192
#define QKV_N   (3 * D_MODEL)     // 2304

typedef __attribute__((ext_vector_type(16))) __bf16 v16bf;
typedef __attribute__((ext_vector_type(8)))  float  v8f;

__device__ __forceinline__ unsigned short f32_to_bf16(float f) {
    unsigned int u = __float_as_uint(f);
    u += 0x7FFFu + ((u >> 16) & 1u);   // round to nearest even
    return (unsigned short)(u >> 16);
}

__device__ __forceinline__ v8f wmma_bf16(v16bf a, v16bf b, v8f c) {
    return __builtin_amdgcn_wmma_f32_16x16x32_bf16(
        /*neg_a=*/false, a, /*neg_b=*/false, b,
        /*c_mod=*/(short)0, c, /*reuse_a=*/false, /*reuse_b=*/false);
}

union FragU { v16bf v; uint4 u[2]; };

// A fragment (16 x K=32, bf16): lane r=lane&15 holds row r; lanes<16 take K
// chunks {0..7,16..23}, lanes>=16 take {8..15,24..31}. stride in halves.
__device__ __forceinline__ v16bf load_frag_a(const unsigned short* p, int stride,
                                             int row0, int col0, int lane) {
    const int r = lane & 15, g = lane >> 4;
    const unsigned short* base = p + (size_t)(row0 + r) * stride + col0 + g * 8;
    FragU f;
    f.u[0] = *(const uint4*)(base);
    f.u[1] = *(const uint4*)(base + 16);
    return f.v;
}

// B fragment (K=32 x 16) supplied from "Bt" storage: row = output column n,
// col = k. Lane n (n<16) holds K {0..7,16..23} of column n, lane n+16 holds K {8..15,24..31}.
__device__ __forceinline__ v16bf load_frag_b(const unsigned short* p, int stride,
                                             int row0, int col0, int lane) {
    const int r = lane & 15, g = lane >> 4;
    const unsigned short* base = p + (size_t)(row0 + r) * stride + col0 + g * 8;
    FragU f;
    f.u[0] = *(const uint4*)(base);
    f.u[1] = *(const uint4*)(base + 16);
    return f.v;
}

// -------- CDNA5 async LDS staging (ASYNCcnt-tracked, bypasses VGPRs) --------
__device__ __forceinline__ unsigned lds_off32(const void* p) {
    // generic LDS address = aperture || offset; low 32 bits are the LDS offset
    return (unsigned)(unsigned long long)p;
}
__device__ __forceinline__ void async_load_b128(unsigned lds_addr, const void* gaddr) {
    asm volatile("global_load_async_to_lds_b128 %0, %1, off"
                 :: "v"(lds_addr), "v"(gaddr) : "memory");
}
__device__ __forceinline__ void wait_async0() {
    asm volatile("s_wait_asynccnt 0" ::: "memory");
}

// ---------------- f32 -> bf16 converter ----------------
__global__ void cvt_bf16_kernel(const float* __restrict__ in,
                                unsigned short* __restrict__ out, int n) {
    int i = blockIdx.x * blockDim.x + threadIdx.x;
    if (i < n) out[i] = f32_to_bf16(in[i]);
}

// ---------------- QKV projection GEMM ----------------
// C[8192 x 2304] = Xb @ Wb^T + bias ; scatter to Q,K (row major [BH,N,64])
// and Vt (transposed [BH,64,N]) as bf16. Double-buffered async LDS staging.
__global__ __launch_bounds__(256) void gemm_qkv_kernel(
    const unsigned short* __restrict__ Xb, const unsigned short* __restrict__ Wb,
    const float* __restrict__ bias, unsigned short* __restrict__ Qb,
    unsigned short* __restrict__ Kb, unsigned short* __restrict__ Vtb) {
    __shared__ __attribute__((aligned(16))) unsigned short As[2][128 * 40];
    __shared__ __attribute__((aligned(16))) unsigned short Bs[2][128 * 40];
    const int tid = threadIdx.x, lane = tid & 31, wave = tid >> 5;
    const int wm = wave & 1, wn = wave >> 1;
    const int nt = QKV_N / 128;                       // 18
    const int m0 = (blockIdx.x / nt) * 128;
    const int n0 = (blockIdx.x % nt) * 128;

    v8f acc[4][2];
#pragma unroll
    for (int i = 0; i < 4; ++i)
#pragma unroll
        for (int j = 0; j < 2; ++j) acc[i][j] = (v8f){};

    // async copy of one 128x32 A tile + 128x32 B tile into buffer `bf`
    auto issue_tiles = [&](int kk, int bf) {
#pragma unroll
        for (int it = 0; it < 2; ++it) {
            const int idx = tid + it * 256;
            const int row = idx >> 2, u = idx & 3;
            async_load_b128(lds_off32(&As[bf][row * 40 + u * 8]),
                            &Xb[(size_t)(m0 + row) * D_MODEL + kk + u * 8]);
            async_load_b128(lds_off32(&Bs[bf][row * 40 + u * 8]),
                            &Wb[(size_t)(n0 + row) * D_MODEL + kk + u * 8]);
        }
    };

    issue_tiles(0, 0);
    int bf = 0;
    for (int kk = 0; kk < D_MODEL; kk += 32) {
        wait_async0();
        __syncthreads();
        if (kk + 32 < D_MODEL) issue_tiles(kk + 32, bf ^ 1);

        v16bf a[4], b[2];
#pragma unroll
        for (int i = 0; i < 4; ++i) a[i] = load_frag_a(As[bf], 40, wm * 64 + i * 16, 0, lane);
#pragma unroll
        for (int j = 0; j < 2; ++j) b[j] = load_frag_b(Bs[bf], 40, wn * 32 + j * 16, 0, lane);
#pragma unroll
        for (int i = 0; i < 4; ++i)
#pragma unroll
            for (int j = 0; j < 2; ++j) acc[i][j] = wmma_bf16(a[i], b[j], acc[i][j]);
        bf ^= 1;
    }

    const int g = lane >> 4, c = lane & 15;
#pragma unroll
    for (int i = 0; i < 4; ++i)
#pragma unroll
        for (int j = 0; j < 2; ++j) {
            const int ncol = n0 + wn * 32 + j * 16 + c;
            const float bv = bias[ncol];
            const int which = ncol / D_MODEL;
            const int rem = ncol - which * D_MODEL;
            const int h = rem >> 6, dd = rem & 63;
#pragma unroll
            for (int jj = 0; jj < 8; ++jj) {
                const int m = m0 + wm * 64 + i * 16 + jj + g * 8;
                const int bb = m >> 12, nr = m & (SEQ - 1);
                const unsigned short val = f32_to_bf16(acc[i][j][jj] + bv);
                const size_t bh = (size_t)(bb * N_HEADS + h);
                if (which == 0)      Qb[(bh * SEQ + nr) * 64 + dd] = val;
                else if (which == 1) Kb[(bh * SEQ + nr) * 64 + dd] = val;
                else                 Vtb[(bh * 64 + dd) * SEQ + nr] = val;
            }
        }
}

// ---------------- Flash attention ----------------
// One block (4 waves) per (b,h, 64-row query tile). Each wave owns 16 q rows.
// K/V tiles double-buffered through async LDS copies.
__global__ __launch_bounds__(128) void attn_kernel(
    const unsigned short* __restrict__ Qb, const unsigned short* __restrict__ Kb,
    const unsigned short* __restrict__ Vtb, unsigned short* __restrict__ Ob) {
    __shared__ __attribute__((aligned(16))) unsigned short Ks[2][64 * 72];
    __shared__ __attribute__((aligned(16))) unsigned short Vs[2][64 * 72];
    __shared__ __attribute__((aligned(16))) unsigned short Ps[4][16 * 72];

    const int tid = threadIdx.x, lane = tid & 31, wave = tid >> 5;
    const int g = lane >> 4, c = lane & 15;
    const int qt = blockIdx.x & 63;      // 4096/64 tiles
    const int bh = blockIdx.x >> 6;      // 0..23
    const size_t base = (size_t)bh * SEQ * 64;
    const int qbase = qt * 64 + wave * 16;

    v16bf qf[2];
#pragma unroll
    for (int kc = 0; kc < 2; ++kc)
        qf[kc] = load_frag_a(Qb + base, 64, qbase, kc * 32, lane);

    v8f o[4];
#pragma unroll
    for (int of = 0; of < 4; ++of) o[of] = (v8f){};
    float mrow[8], lrow[8];
#pragma unroll
    for (int jj = 0; jj < 8; ++jj) { mrow[jj] = -3.0e38f; lrow[jj] = 0.f; }

    // async copy of K tile [64 x 64] and Vt tile [64 x 64] for key block kt
    auto issue_kv = [&](int kt, int bf) {
        const int row = tid >> 1, half = tid & 1;
#pragma unroll
        for (int q4 = 0; q4 < 4; ++q4) {
            async_load_b128(lds_off32(&Ks[bf][row * 72 + half * 32 + q4 * 8]),
                            Kb + base + (size_t)(kt * 64 + row) * 64 + half * 32 + q4 * 8);
            async_load_b128(lds_off32(&Vs[bf][row * 72 + half * 32 + q4 * 8]),
                            Vtb + base + (size_t)row * SEQ + kt * 64 + half * 32 + q4 * 8);
        }
    };

    issue_kv(0, 0);
    int bf = 0;
    for (int kt = 0; kt < SEQ / 64; ++kt) {
        wait_async0();
        __syncthreads();
        if (kt + 1 < SEQ / 64) issue_kv(kt + 1, bf ^ 1);

        // S = Q @ K^T  (16 x 64), fp32 accum
        v8f s[4];
#pragma unroll
        for (int nf = 0; nf < 4; ++nf) s[nf] = (v8f){};
#pragma unroll
        for (int kc = 0; kc < 2; ++kc) {
            v16bf aq = qf[kc];
#pragma unroll
            for (int nf = 0; nf < 4; ++nf)
                s[nf] = wmma_bf16(aq, load_frag_b(Ks[bf], 72, nf * 16, kc * 32, lane), s[nf]);
        }
#pragma unroll
        for (int nf = 0; nf < 4; ++nf)
#pragma unroll
            for (int jj = 0; jj < 8; ++jj) s[nf][jj] *= 0.125f;  // 1/sqrt(64)

        // online softmax; row jj+g*8 spans the 16 lanes of this half-wave
#pragma unroll
        for (int jj = 0; jj < 8; ++jj) {
            float mx = fmaxf(fmaxf(s[0][jj], s[1][jj]), fmaxf(s[2][jj], s[3][jj]));
#pragma unroll
            for (int off = 1; off < 16; off <<= 1) mx = fmaxf(mx, __shfl_xor(mx, off, 32));
            const float mnew = fmaxf(mrow[jj], mx);
            const float corr = __expf(mrow[jj] - mnew);
            mrow[jj] = mnew;
            float ps = 0.f;
#pragma unroll
            for (int nf = 0; nf < 4; ++nf) {
                const float p = __expf(s[nf][jj] - mnew);
                s[nf][jj] = p;
                ps += p;
            }
#pragma unroll
            for (int off = 1; off < 16; off <<= 1) ps += __shfl_xor(ps, off, 32);
            lrow[jj] = lrow[jj] * corr + ps;
#pragma unroll
            for (int of = 0; of < 4; ++of) o[of][jj] *= corr;
        }

        // spill P (bf16) into wave-private LDS for use as WMMA A fragments
#pragma unroll
        for (int nf = 0; nf < 4; ++nf)
#pragma unroll
            for (int jj = 0; jj < 8; ++jj)
                Ps[wave][(jj + g * 8) * 72 + nf * 16 + c] = f32_to_bf16(s[nf][jj]);

        // O += P @ V
#pragma unroll
        for (int kc = 0; kc < 2; ++kc) {
            v16bf ap = load_frag_a(Ps[wave], 72, 0, kc * 32, lane);
#pragma unroll
            for (int of = 0; of < 4; ++of)
                o[of] = wmma_bf16(ap, load_frag_b(Vs[bf], 72, of * 16, kc * 32, lane), o[of]);
        }
        bf ^= 1;
    }

    const int hh = bh % N_HEADS, bb = bh / N_HEADS;
#pragma unroll
    for (int of = 0; of < 4; ++of)
#pragma unroll
        for (int jj = 0; jj < 8; ++jj) {
            const int q = qbase + jj + g * 8;
            const int col = hh * 64 + of * 16 + c;
            Ob[((size_t)(bb * SEQ + q)) * D_MODEL + col] = f32_to_bf16(o[of][jj] / lrow[jj]);
        }
}

// ---------------- Output projection GEMM ----------------
__global__ __launch_bounds__(256) void gemm_out_kernel(
    const unsigned short* __restrict__ Ab, const unsigned short* __restrict__ Wb,
    const float* __restrict__ bias, float* __restrict__ out) {
    __shared__ __attribute__((aligned(16))) unsigned short As[2][128 * 40];
    __shared__ __attribute__((aligned(16))) unsigned short Bs[2][128 * 40];
    const int tid = threadIdx.x, lane = tid & 31, wave = tid >> 5;
    const int wm = wave & 1, wn = wave >> 1;
    const int nt = D_MODEL / 128;                    // 6
    const int m0 = (blockIdx.x / nt) * 128;
    const int n0 = (blockIdx.x % nt) * 128;

    v8f acc[4][2];
#pragma unroll
    for (int i = 0; i < 4; ++i)
#pragma unroll
        for (int j = 0; j < 2; ++j) acc[i][j] = (v8f){};

    auto issue_tiles = [&](int kk, int bf) {
#pragma unroll
        for (int it = 0; it < 2; ++it) {
            const int idx = tid + it * 256;
            const int row = idx >> 2, u = idx & 3;
            async_load_b128(lds_off32(&As[bf][row * 40 + u * 8]),
                            &Ab[(size_t)(m0 + row) * D_MODEL + kk + u * 8]);
            async_load_b128(lds_off32(&Bs[bf][row * 40 + u * 8]),
                            &Wb[(size_t)(n0 + row) * D_MODEL + kk + u * 8]);
        }
    };

    issue_tiles(0, 0);
    int bf = 0;
    for (int kk = 0; kk < D_MODEL; kk += 32) {
        wait_async0();
        __syncthreads();
        if (kk + 32 < D_MODEL) issue_tiles(kk + 32, bf ^ 1);

        v16bf a[4], b[2];
#pragma unroll
        for (int i = 0; i < 4; ++i) a[i] = load_frag_a(As[bf], 40, wm * 64 + i * 16, 0, lane);
#pragma unroll
        for (int j = 0; j < 2; ++j) b[j] = load_frag_b(Bs[bf], 40, wn * 32 + j * 16, 0, lane);
#pragma unroll
        for (int i = 0; i < 4; ++i)
#pragma unroll
            for (int j = 0; j < 2; ++j) acc[i][j] = wmma_bf16(a[i], b[j], acc[i][j]);
        bf ^= 1;
    }

    const int g = lane >> 4, c = lane & 15;
#pragma unroll
    for (int i = 0; i < 4; ++i)
#pragma unroll
        for (int j = 0; j < 2; ++j) {
            const int ncol = n0 + wn * 32 + j * 16 + c;
            const float bv = bias[ncol];
#pragma unroll
            for (int jj = 0; jj < 8; ++jj) {
                const int m = m0 + wm * 64 + i * 16 + jj + g * 8;
                out[(size_t)m * D_MODEL + ncol] = acc[i][j][jj] + bv;
            }
        }
}

extern "C" void kernel_launch(void* const* d_in, const int* in_sizes, int n_in,
                              void* d_out, int out_size, void* d_ws, size_t ws_size,
                              hipStream_t stream) {
    const float* x      = (const float*)d_in[0];
    const float* qkv_w  = (const float*)d_in[1];
    const float* qkv_b  = (const float*)d_in[2];
    const float* out_w  = (const float*)d_in[3];
    const float* out_b  = (const float*)d_in[4];
    float* out = (float*)d_out;

    char* ws = (char*)d_ws;
    auto alloc = [&](size_t elems) {
        unsigned short* p = (unsigned short*)ws;
        ws += (elems * sizeof(unsigned short) + 255) & ~(size_t)255;
        return p;
    };
    const size_t nX  = (size_t)ROWS * D_MODEL;
    const size_t nWq = (size_t)QKV_N * D_MODEL;
    const size_t nWo = (size_t)D_MODEL * D_MODEL;
    const size_t nHd = (size_t)BATCH * N_HEADS * SEQ * D_HEAD;
    unsigned short* Xb  = alloc(nX);
    unsigned short* Wqb = alloc(nWq);
    unsigned short* Wob = alloc(nWo);
    unsigned short* Qb  = alloc(nHd);
    unsigned short* Kb  = alloc(nHd);
    unsigned short* Vtb = alloc(nHd);
    unsigned short* Ob  = alloc(nX);

    cvt_bf16_kernel<<<(int)((nX  + 255) / 256), 256, 0, stream>>>(x,     Xb,  (int)nX);
    cvt_bf16_kernel<<<(int)((nWq + 255) / 256), 256, 0, stream>>>(qkv_w, Wqb, (int)nWq);
    cvt_bf16_kernel<<<(int)((nWo + 255) / 256), 256, 0, stream>>>(out_w, Wob, (int)nWo);

    gemm_qkv_kernel<<<(ROWS / 128) * (QKV_N / 128), 256, 0, stream>>>(
        Xb, Wqb, qkv_b, Qb, Kb, Vtb);

    attn_kernel<<<BATCH * N_HEADS * (SEQ / 64), 128, 0, stream>>>(Qb, Kb, Vtb, Ob);

    gemm_out_kernel<<<(ROWS / 128) * (D_MODEL / 128), 256, 0, stream>>>(
        Ob, Wob, out_b, out);
}
